// UniTransformerO2TwoUpdateGeneral_22857815949681
// MI455X (gfx1250) — matrix operand order repaired and checked by reading, not compiled, not run on hardware
//
#include <hip/hip_runtime.h>
#include <math.h>

// ---------------------------------------------------------------------------
// UniTransformerO2TwoUpdateGeneral for MI455X (gfx1250, wave32, WMMA)
// Per-edge MLPs (340->128->128 / 340->128->16) are the FLOP hot spot
// (~1.4e11 FLOPs) -> v_wmma_f32_16x16x32_f16 with LDS-staged weights and
// gathered kv tiles. Staging uses global_load_async_to_lds_b128 (ASYNCcnt).
// kv feature order is permuted to [h_dst | h_src | edge_attr | r_feat | pad]
// (and W1^T rows permuted identically) so every gather is 16B-aligned.
// Elementwise/softmax/scatter stages are bandwidth-trivial vs 23.3 TB/s.
// ---------------------------------------------------------------------------

#define USE_ASYNC 1

typedef _Float16 half_t;
typedef __attribute__((ext_vector_type(16))) _Float16 v16h;
typedef __attribute__((ext_vector_type(8)))  _Float16 v8h;
typedef __attribute__((ext_vector_type(8)))  float    v8f;

#define N_HEADS 16
#define HD      8
#define HID     128
#define KVDIM   340
#define KVPAD   352
#define RFEAT   80

static __device__ inline v8f vzero8() {
  v8f r;
#pragma unroll
  for (int i = 0; i < 8; ++i) r[i] = 0.0f;
  return r;
}

static __device__ inline v16h combine16(v8h a, v8h b) {
  v16h r;
#pragma unroll
  for (int i = 0; i < 8; ++i) { r[i] = a[i]; r[i + 8] = b[i]; }
  return r;
}

// A-matrix 16x32 f16 fragment (ISA 7.12.2): lane = (hf*16 + m), per lane
// k in {hf*8..hf*8+7} (VGPR0-3) and {16+hf*8..+7} (VGPR4-7) -> two b128 loads.
static __device__ inline v16h load_afrag(const half_t* row, int ks, int hf) {
  const v8h a0 = *(const v8h*)(row + ks + hf * 8);
  const v8h a1 = *(const v8h*)(row + ks + 16 + hf * 8);
  return combine16(a0, a1);
}

// B-matrix 32x16 f16 fragment: lane = (hf*16 + n); per lane k in
// {hf*8..hf*8+7} and {16+hf*8..+7}. B stored transposed: rows of W^T are contiguous in k.
static __device__ inline v16h load_bfrag(const half_t* row, int ks, int hf) {
  const v8h b0 = *(const v8h*)(row + ks + hf * 8);
  const v8h b1 = *(const v8h*)(row + ks + 16 + hf * 8);
  return combine16(b0, b1);
}

// 16-byte global -> LDS copy. Async path uses the CDNA5 async-DMA unit
// (ASYNCcnt); fallback is a regular VGPR round trip.
static __device__ inline void lds_copy16(void* lds_dst, const void* gsrc) {
#if USE_ASYNC
  unsigned off = (unsigned)(uintptr_t)lds_dst;  // generic LDS ptr low 32 = offset
  asm volatile("global_load_async_to_lds_b128 %0, %1, off"
               :: "v"(off), "v"(gsrc) : "memory");
#else
  *(v8h*)lds_dst = *(const v8h*)gsrc;
#endif
}
static __device__ inline void async_fence() {
#if USE_ASYNC
  asm volatile("s_wait_asynccnt 0x0" ::: "memory");
#endif
}

// ---------------------------------------------------------------------------
// Fused 2-layer MLP (Linear -> LayerNorm -> ReLU -> Linear) with WMMA.
// 128 threads = 4 waves; each workgroup handles 64 rows.
// mode 0: rows come from a16 [R][DINP] (node features)
// mode 1: rows are gathered kv_in (permuted layout, DINP must be KVPAD)
// ---------------------------------------------------------------------------
template <int DINP, int DOUT>
__global__ __launch_bounds__(128) void mlp_fused(
    int R, int mode,
    const half_t* __restrict__ a16,
    const float* __restrict__ edge_attr,
    const half_t* __restrict__ rfeat16,
    const half_t* __restrict__ hnode16,
    const int* __restrict__ srcIdx,
    const int* __restrict__ dstIdx,
    const half_t* __restrict__ w1t,   // [128][DINP]  (W1^T, f16, k-permuted/padded)
    const float* __restrict__ b1,     // [128]
    const float* __restrict__ gamma,
    const float* __restrict__ beta,
    const half_t* __restrict__ w2t,   // [DOUT][128]  (W2^T, f16) - read from global
    const float* __restrict__ b2,     // [DOUT]
    const float* __restrict__ ew,     // nullable [R] post-scale
    const float* __restrict__ resid,  // nullable [R][DOUT] residual add
    half_t* __restrict__ out16,       // nullable [R][DOUT]
    float* __restrict__ out32)        // nullable [R][DOUT]
{
  constexpr int W1_BYTES = 128 * DINP * 2;
  constexpr int A_BYTES  = 64 * DINP * 2;
  constexpr int Z_BYTES  = 64 * 128 * 4;
  constexpr int C_BYTES  = (A_BYTES > Z_BYTES) ? A_BYTES : Z_BYTES;
  constexpr int Z16_BYTES = 64 * 128 * 2;
  __shared__ __align__(16) char smem[W1_BYTES + C_BYTES + Z16_BYTES];

  half_t* W1s = (half_t*)smem;
  half_t* As  = (half_t*)(smem + W1_BYTES);        // overlaid with Zs
  float*  Zs  = (float*)(smem + W1_BYTES);
  half_t* Z16 = (half_t*)(smem + W1_BYTES + C_BYTES);

  const int tid  = threadIdx.x;
  const int base = blockIdx.x * 64;

  // ---- Stage W1^T into LDS in 16B chunks (async DMA) ----
  for (int i = tid; i < 128 * DINP / 8; i += 128)
    lds_copy16(W1s + i * 8, w1t + i * 8);

  // ---- Stage A tile (64 rows x DINP) ----
  if (mode == 0) {
    constexpr int CPR = DINP / 8;  // chunks per row
    for (int i = tid; i < 64 * CPR; i += 128) {
      int r = i / CPR, part = i - r * CPR;
      int row = base + r; if (row >= R) row = R - 1;
      lds_copy16(As + r * DINP + part * 8, a16 + (long)row * DINP + part * 8);
    }
  } else if constexpr (DINP == KVPAD) {
    // cols [0,128)=h[dst], [128,256)=h[src]: 16B-aligned gather-to-LDS
    for (int ch = tid; ch < 64 * 32; ch += 128) {
      int r = ch >> 5, rem = ch & 31;
      int which = rem >> 4, part = rem & 15;
      int e = base + r; if (e >= R) e = R - 1;
      int node = which ? srcIdx[e] : dstIdx[e];
      lds_copy16(As + r * DINP + which * 128 + part * 8,
                 hnode16 + (long)node * HID + part * 8);
    }
    // cols [256,260)=edge_attr, [260,340)=r_feat, [340,352)=pad : u32 path
    const unsigned* rf32 = (const unsigned*)rfeat16;
    for (int i = tid; i < 64 * 48; i += 128) {
      int r = i / 48, w = i - r * 48;
      int e = base + r; if (e >= R) e = R - 1;
      unsigned val;
      if (w < 2) {
        union { half_t h[2]; unsigned u; } u2;
        u2.h[0] = (half_t)edge_attr[(long)e * 4 + 2 * w];
        u2.h[1] = (half_t)edge_attr[(long)e * 4 + 2 * w + 1];
        val = u2.u;
      } else if (w < 42) {
        val = rf32[(long)e * 40 + (w - 2)];
      } else {
        val = 0u;
      }
      *((unsigned*)(As + r * DINP + 256) + w) = val;
    }
  }
  async_fence();
  __syncthreads();

  const int lane = tid & 31;
  const int wv   = tid >> 5;      // wave id: rows [wv*16, wv*16+16)
  const int lm   = lane & 15;
  const int hf   = lane >> 4;

  // ---- GEMM 1: z = A @ W1  (16 rows x 128 cols per wave) ----
  v8f acc[8];
#pragma unroll
  for (int c = 0; c < 8; ++c) acc[c] = vzero8();

  const half_t* arow = As + (wv * 16 + lm) * DINP;
#pragma unroll 2
  for (int ks = 0; ks < DINP; ks += 32) {
    v16h af = load_afrag(arow, ks, hf);
#pragma unroll
    for (int c = 0; c < 8; ++c) {
      v16h bf = load_bfrag(W1s + (c * 16 + lm) * DINP, ks, hf);
      acc[c] = __builtin_amdgcn_wmma_f32_16x16x32_f16(
          false, af, false, bf, (short)0, acc[c], false, false);
    }
  }
  __syncthreads();  // done reading As before overlaying Zs

  // Store z + bias to LDS (D layout: vgpr i -> m = hf*8+i, lane -> n).
#pragma unroll
  for (int c = 0; c < 8; ++c) {
    int col = c * 16 + lm;
    float bb = b1[col];
#pragma unroll
    for (int i = 0; i < 8; ++i)
      Zs[(wv * 16 + hf * 8 + i) * 128 + col] = acc[c][i] + bb;
  }
  __syncthreads();

  // ---- LayerNorm + ReLU per row -> f16 ----
  if (tid < 64) {
    const float* zr = Zs + tid * 128;
    float mu = 0.f;
#pragma unroll 8
    for (int j = 0; j < 128; ++j) mu += zr[j];
    mu *= (1.0f / 128.0f);
    float var = 0.f;
#pragma unroll 8
    for (int j = 0; j < 128; ++j) { float d = zr[j] - mu; var += d * d; }
    var *= (1.0f / 128.0f);
    float rstd = rsqrtf(var + 1e-5f);
    for (int j = 0; j < 128; ++j) {
      float v = (zr[j] - mu) * rstd * gamma[j] + beta[j];
      Z16[tid * 128 + j] = (half_t)(v > 0.f ? v : 0.f);
    }
  }
  __syncthreads();

  // ---- GEMM 2: out = relu(z) @ W2 ----
  constexpr int CT2 = DOUT / 16;
  v8f acc2[CT2];
#pragma unroll
  for (int c = 0; c < CT2; ++c) acc2[c] = vzero8();

  const half_t* zrow = Z16 + (wv * 16 + lm) * 128;
#pragma unroll 2
  for (int ks = 0; ks < 128; ks += 32) {
    v16h af = load_afrag(zrow, ks, hf);
#pragma unroll
    for (int c = 0; c < CT2; ++c) {
      v16h bf = load_bfrag(w2t + (c * 16 + lm) * 128, ks, hf);
      acc2[c] = __builtin_amdgcn_wmma_f32_16x16x32_f16(
          false, af, false, bf, (short)0, acc2[c], false, false);
    }
  }

  // ---- epilogue: bias, optional e_w scale / residual, store ----
#pragma unroll
  for (int c = 0; c < CT2; ++c) {
    int col = c * 16 + lm;
    float bb = b2[col];
#pragma unroll
    for (int i = 0; i < 8; ++i) {
      int gr = base + wv * 16 + hf * 8 + i;
      if (gr < R) {
        float v = acc2[c][i] + bb;
        if (ew)    v *= ew[gr];
        if (resid) v += resid[(long)gr * DOUT + col];
        if (out16) out16[(long)gr * DOUT + col] = (half_t)v;
        if (out32) out32[(long)gr * DOUT + col] = v;
      }
    }
  }
}

// ---------------------------------------------------------------------------
// Helper kernels
// ---------------------------------------------------------------------------
__global__ __launch_bounds__(256) void fill_u32(unsigned* p, unsigned v, long n) {
  long i = (long)blockIdx.x * 256 + threadIdx.x;
  if (i < n) p[i] = v;
}

__global__ __launch_bounds__(256) void f32_to_f16(const float* s, half_t* d, long n) {
  long i = (long)blockIdx.x * 256 + threadIdx.x;
  if (i < n) d[i] = (half_t)s[i];
}

// out[n][k'] = w1^T with optional kv permutation + f16 + K-padding.
// kvperm: k' in [0,256) -> orig 84+k' (h_dst|h_src), k' in [256,340) -> k'-256
__global__ __launch_bounds__(256) void wt_kernel(const float* w, half_t* o,
                                                 int K, int Ncols, int Kpad,
                                                 int kvperm) {
  int i = blockIdx.x * 256 + threadIdx.x;
  if (i >= Ncols * Kpad) return;
  int n = i / Kpad, k = i - n * Kpad;
  half_t v = (half_t)0.0f;
  if (k < K) {
    int ko = k;
    if (kvperm) ko = (k < 256) ? (84 + k) : (k - 256);
    v = (half_t)w[ko * Ncols + n];
  }
  o[i] = v;
}

__global__ __launch_bounds__(256) void edge_prep(
    const float* __restrict__ x, const float* __restrict__ ea,
    const int* __restrict__ src, const int* __restrict__ dst,
    const float* __restrict__ eww1, const float* __restrict__ ewb1,
    const float* __restrict__ eww2, const float* __restrict__ ewb2,
    float* __restrict__ relx, half_t* __restrict__ rfeat,
    float* __restrict__ ew1o, float* __restrict__ ew2o, int E) {
  int e = blockIdx.x * 256 + threadIdx.x;
  if (e >= E) return;
  int s = src[e], d = dst[e];
  float rx = x[d * 3 + 0] - x[s * 3 + 0];
  float ry = x[d * 3 + 1] - x[s * 3 + 1];
  float rz = x[d * 3 + 2] - x[s * 3 + 2];
  relx[e * 3 + 0] = rx; relx[e * 3 + 1] = ry; relx[e * 3 + 2] = rz;
  float dist = sqrtf(rx * rx + ry * ry + rz * rz);
  const float step  = 10.0f / 19.0f;
  const float coeff = -0.5f / (step * step);
  float df[20];
#pragma unroll
  for (int g = 0; g < 20; ++g) {
    float t = dist - step * (float)g;
    df[g] = expf(coeff * t * t);
  }
  float s1 = 0.f, s2 = 0.f;
#pragma unroll
  for (int f = 0; f < 4; ++f) {
    float a = ea[(long)e * 4 + f];
#pragma unroll
    for (int g = 0; g < 20; ++g) {
      float rv = a * df[g];
      rfeat[(long)e * RFEAT + f * 20 + g] = (half_t)rv;
      s1 += rv * eww1[f * 20 + g];
      s2 += rv * eww2[f * 20 + g];
    }
  }
  ew1o[e] = 1.0f / (1.0f + expf(-(s1 + ewb1[0])));
  ew2o[e] = 1.0f / (1.0f + expf(-(s2 + ewb2[0])));
}

__global__ __launch_bounds__(256) void scores_kernel(
    const float* __restrict__ q, const half_t* __restrict__ k,
    const int* __restrict__ dst, float* __restrict__ sc, long total) {
  long t = (long)blockIdx.x * 256 + threadIdx.x;
  if (t >= total) return;
  long e = t >> 4; int hh = (int)(t & 15);
  const float*  qp = q + (long)dst[e] * HID + hh * HD;
  const half_t* kp = k + e * HID + hh * HD;
  float s = 0.f;
#pragma unroll
  for (int j = 0; j < HD; ++j) s += qp[j] * (float)kp[j];
  sc[t] = s * 0.35355339059327373f;  // 1/sqrt(8)
}

static __device__ inline unsigned fkey(float f) {
  unsigned b = __float_as_uint(f);
  return (b & 0x80000000u) ? ~b : (b | 0x80000000u);
}
static __device__ inline float funkey(unsigned u) {
  return (u & 0x80000000u) ? __uint_as_float(u ^ 0x80000000u)
                           : __uint_as_float(~u);
}

__global__ __launch_bounds__(256) void segmax_kernel(
    const float* __restrict__ sc, const int* __restrict__ dst,
    unsigned* __restrict__ mk, long total) {
  long t = (long)blockIdx.x * 256 + threadIdx.x;
  if (t >= total) return;
  long e = t >> 4; int hh = (int)(t & 15);
  atomicMax(&mk[(long)dst[e] * N_HEADS + hh], fkey(sc[t]));
}

__global__ __launch_bounds__(256) void expsum_kernel(
    const float* __restrict__ sc, const int* __restrict__ dst,
    const unsigned* __restrict__ mk, float* __restrict__ esc,
    float* __restrict__ sum, long total) {
  long t = (long)blockIdx.x * 256 + threadIdx.x;
  if (t >= total) return;
  long e = t >> 4; int hh = (int)(t & 15);
  float m = funkey(mk[(long)dst[e] * N_HEADS + hh]);
  float ev = expf(sc[t] - m);
  esc[t] = ev;
  atomicAdd(&sum[(long)dst[e] * N_HEADS + hh], ev);
}

__global__ __launch_bounds__(256) void agg_x2h_kernel(
    const float* __restrict__ esc, const float* __restrict__ sum,
    const half_t* __restrict__ v, const int* __restrict__ dst,
    float* __restrict__ agg, long total) {
  long t = (long)blockIdx.x * 256 + threadIdx.x;
  if (t >= total) return;
  long e = t >> 7; int j = (int)(t & 127); int hh = j >> 3;
  int d = dst[e];
  float alpha = esc[e * N_HEADS + hh] / sum[(long)d * N_HEADS + hh];
  atomicAdd(&agg[(long)d * HID + j], alpha * (float)v[t]);
}

__global__ __launch_bounds__(256) void nodecat_kernel(
    const float* __restrict__ agg, const float* __restrict__ h,
    half_t* __restrict__ cat, long total) {
  long t = (long)blockIdx.x * 256 + threadIdx.x;
  if (t >= total) return;
  long n = t >> 8; int c = (int)(t & 255);
  float v = (c < 128) ? agg[n * HID + c] : h[n * HID + (c - 128)];
  cat[t] = (half_t)v;
}

__global__ __launch_bounds__(256) void agg_h2x_kernel(
    const float* __restrict__ esc, const float* __restrict__ sum,
    const half_t* __restrict__ v2, const float* __restrict__ relx,
    const int* __restrict__ dst, float* __restrict__ delta, long total) {
  long t = (long)blockIdx.x * 256 + threadIdx.x;
  if (t >= total) return;
  long e = t >> 4; int hh = (int)(t & 15);
  int d = dst[e];
  float alpha = esc[t] / sum[(long)d * N_HEADS + hh];
  float vv = alpha * (float)v2[t];
  atomicAdd(&delta[d * 3 + 0], vv * relx[e * 3 + 0]);
  atomicAdd(&delta[d * 3 + 1], vv * relx[e * 3 + 1]);
  atomicAdd(&delta[d * 3 + 2], vv * relx[e * 3 + 2]);
}

__global__ __launch_bounds__(256) void finalx_kernel(
    const float* __restrict__ x, const float* __restrict__ delta,
    const int* __restrict__ mask, float* __restrict__ out, long total) {
  long t = (long)blockIdx.x * 256 + threadIdx.x;
  if (t >= total) return;
  long n = t / 3;
  out[t] = x[t] + (delta[t] * (1.0f / (float)N_HEADS)) * (float)mask[n];
}

// ---------------------------------------------------------------------------
// Host launcher
// ---------------------------------------------------------------------------
extern "C" void kernel_launch(void* const* d_in, const int* in_sizes, int n_in,
                              void* d_out, int out_size, void* d_ws, size_t ws_size,
                              hipStream_t stream) {
  (void)n_in; (void)out_size;
  const float* h   = (const float*)d_in[0];
  const float* x   = (const float*)d_in[1];
  const float* ea  = (const float*)d_in[2];
  const int*   eix = (const int*)d_in[3];   // edge_index [2,E] (int32 per harness)
  const int*   msk = (const int*)d_in[4];

  const int E = in_sizes[2] / 4;
  const int N = in_sizes[0] / HID;
  const int* src = eix;
  const int* dst = eix + E;

  // params (setup_inputs dict order, MLPs flattened w1,b1,g,be,w2,b2)
  const float* hk_w1 = (const float*)d_in[5];  const float* hk_b1 = (const float*)d_in[6];
  const float* hk_g  = (const float*)d_in[7];  const float* hk_be = (const float*)d_in[8];
  const float* hk_w2 = (const float*)d_in[9];  const float* hk_b2 = (const float*)d_in[10];
  const float* hv_w1 = (const float*)d_in[11]; const float* hv_b1 = (const float*)d_in[12];
  const float* hv_g  = (const float*)d_in[13]; const float* hv_be = (const float*)d_in[14];
  const float* hv_w2 = (const float*)d_in[15]; const float* hv_b2 = (const float*)d_in[16];
  const float* hq_w1 = (const float*)d_in[17]; const float* hq_b1 = (const float*)d_in[18];
  const float* hq_g  = (const float*)d_in[19]; const float* hq_be = (const float*)d_in[20];
  const float* hq_w2 = (const float*)d_in[21]; const float* hq_b2 = (const float*)d_in[22];
  const float* ew1w  = (const float*)d_in[23]; const float* ew1b  = (const float*)d_in[24];
  const float* no_w1 = (const float*)d_in[25]; const float* no_b1 = (const float*)d_in[26];
  const float* no_g  = (const float*)d_in[27]; const float* no_be = (const float*)d_in[28];
  const float* no_w2 = (const float*)d_in[29]; const float* no_b2 = (const float*)d_in[30];
  const float* xk_w1 = (const float*)d_in[31]; const float* xk_b1 = (const float*)d_in[32];
  const float* xk_g  = (const float*)d_in[33]; const float* xk_be = (const float*)d_in[34];
  const float* xk_w2 = (const float*)d_in[35]; const float* xk_b2 = (const float*)d_in[36];
  const float* xv_w1 = (const float*)d_in[37]; const float* xv_b1 = (const float*)d_in[38];
  const float* xv_g  = (const float*)d_in[39]; const float* xv_be = (const float*)d_in[40];
  const float* xv_w2 = (const float*)d_in[41]; const float* xv_b2 = (const float*)d_in[42];
  const float* xq_w1 = (const float*)d_in[43]; const float* xq_b1 = (const float*)d_in[44];
  const float* xq_g  = (const float*)d_in[45]; const float* xq_be = (const float*)d_in[46];
  const float* xq_w2 = (const float*)d_in[47]; const float* xq_b2 = (const float*)d_in[48];
  const float* ew2w  = (const float*)d_in[49]; const float* ew2b  = (const float*)d_in[50];

  // ---- carve workspace (requires ~290 MB) ----
  char* ws = (char*)d_ws; size_t off = 0; (void)ws_size;
  auto alloc = [&](size_t bytes) -> char* {
    char* p = ws + off; off += (bytes + 255) & ~(size_t)255; return p;
  };
  half_t* h16     = (half_t*)alloc((size_t)N * HID * 2);
  half_t* newh16  = (half_t*)alloc((size_t)N * HID * 2);
  half_t* rfeat16 = (half_t*)alloc((size_t)E * RFEAT * 2);
  float*  ew1     = (float*)alloc((size_t)E * 4);
  float*  ew2     = (float*)alloc((size_t)E * 4);
  float*  relx    = (float*)alloc((size_t)E * 3 * 4);
  half_t* kbuf    = (half_t*)alloc((size_t)E * HID * 2);
  half_t* vbuf    = (half_t*)alloc((size_t)E * HID * 2);   // reused as v2 [E,16]
  float*  qbuf    = (float*)alloc((size_t)N * HID * 4);
  float*  scbuf   = (float*)alloc((size_t)E * N_HEADS * 4);
  float*  escbuf  = (float*)alloc((size_t)E * N_HEADS * 4);
  unsigned* mkbuf = (unsigned*)alloc((size_t)N * N_HEADS * 4);
  float*  sumbuf  = (float*)alloc((size_t)N * N_HEADS * 4);
  float*  aggbuf  = (float*)alloc((size_t)N * HID * 4);
  half_t* catbuf  = (half_t*)alloc((size_t)N * 2 * HID * 2);
  float*  delta   = (float*)alloc((size_t)N * 3 * 4);
  half_t* hk_w1t  = (half_t*)alloc(128 * KVPAD * 2);
  half_t* hv_w1t  = (half_t*)alloc(128 * KVPAD * 2);
  half_t* xk_w1t  = (half_t*)alloc(128 * KVPAD * 2);
  half_t* xv_w1t  = (half_t*)alloc(128 * KVPAD * 2);
  half_t* hq_w1t  = (half_t*)alloc(128 * 128 * 2);
  half_t* xq_w1t  = (half_t*)alloc(128 * 128 * 2);
  half_t* no_w1t  = (half_t*)alloc(128 * 256 * 2);
  half_t* hk_w2t  = (half_t*)alloc(128 * 128 * 2);
  half_t* hv_w2t  = (half_t*)alloc(128 * 128 * 2);
  half_t* xk_w2t  = (half_t*)alloc(128 * 128 * 2);
  half_t* xv_w2t  = (half_t*)alloc(16 * 128 * 2);
  half_t* hq_w2t  = (half_t*)alloc(128 * 128 * 2);
  half_t* xq_w2t  = (half_t*)alloc(128 * 128 * 2);
  half_t* no_w2t  = (half_t*)alloc(128 * 128 * 2);

  float* out_h = (float*)d_out;                    // x2h_out [N,128]
  float* out_x = (float*)d_out + (size_t)N * HID;  // x_new [N,3]

  auto cdiv = [](long a, long b) { return (int)((a + b - 1) / b); };
  const int etiles = cdiv(E, 64), ntiles = cdiv(N, 64);

  // ---- weight preprocessing (transpose + f16 + kv-permute + K-pad) ----
  wt_kernel<<<cdiv(128 * KVPAD, 256), 256, 0, stream>>>(hk_w1, hk_w1t, KVDIM, 128, KVPAD, 1);
  wt_kernel<<<cdiv(128 * KVPAD, 256), 256, 0, stream>>>(hv_w1, hv_w1t, KVDIM, 128, KVPAD, 1);
  wt_kernel<<<cdiv(128 * KVPAD, 256), 256, 0, stream>>>(xk_w1, xk_w1t, KVDIM, 128, KVPAD, 1);
  wt_kernel<<<cdiv(128 * KVPAD, 256), 256, 0, stream>>>(xv_w1, xv_w1t, KVDIM, 128, KVPAD, 1);
  wt_kernel<<<cdiv(128 * 128, 256), 256, 0, stream>>>(hq_w1, hq_w1t, 128, 128, 128, 0);
  wt_kernel<<<cdiv(128 * 128, 256), 256, 0, stream>>>(xq_w1, xq_w1t, 128, 128, 128, 0);
  wt_kernel<<<cdiv(128 * 256, 256), 256, 0, stream>>>(no_w1, no_w1t, 256, 128, 256, 0);
  wt_kernel<<<cdiv(128 * 128, 256), 256, 0, stream>>>(hk_w2, hk_w2t, 128, 128, 128, 0);
  wt_kernel<<<cdiv(128 * 128, 256), 256, 0, stream>>>(hv_w2, hv_w2t, 128, 128, 128, 0);
  wt_kernel<<<cdiv(128 * 128, 256), 256, 0, stream>>>(xk_w2, xk_w2t, 128, 128, 128, 0);
  wt_kernel<<<cdiv(16 * 128, 256), 256, 0, stream>>>(xv_w2, xv_w2t, 128, 16, 128, 0);
  wt_kernel<<<cdiv(128 * 128, 256), 256, 0, stream>>>(hq_w2, hq_w2t, 128, 128, 128, 0);
  wt_kernel<<<cdiv(128 * 128, 256), 256, 0, stream>>>(xq_w2, xq_w2t, 128, 128, 128, 0);
  wt_kernel<<<cdiv(128 * 128, 256), 256, 0, stream>>>(no_w2, no_w2t, 128, 128, 128, 0);

  f32_to_f16<<<cdiv((long)N * HID, 256), 256, 0, stream>>>(h, h16, (long)N * HID);
  edge_prep<<<cdiv(E, 256), 256, 0, stream>>>(x, ea, src, dst, ew1w, ew1b, ew2w, ew2b,
                                              relx, rfeat16, ew1, ew2, E);

  // ============================ x2h layer ============================
  mlp_fused<128, 128><<<ntiles, 128, 0, stream>>>(
      N, 0, h16, nullptr, nullptr, nullptr, nullptr, nullptr,
      hq_w1t, hq_b1, hq_g, hq_be, hq_w2t, hq_b2,
      nullptr, nullptr, nullptr, qbuf);
  mlp_fused<KVPAD, 128><<<etiles, 128, 0, stream>>>(
      E, 1, nullptr, ea, rfeat16, h16, src, dst,
      hk_w1t, hk_b1, hk_g, hk_be, hk_w2t, hk_b2,
      nullptr, nullptr, kbuf, nullptr);
  mlp_fused<KVPAD, 128><<<etiles, 128, 0, stream>>>(
      E, 1, nullptr, ea, rfeat16, h16, src, dst,
      hv_w1t, hv_b1, hv_g, hv_be, hv_w2t, hv_b2,
      ew1, nullptr, vbuf, nullptr);

  long nh = (long)N * N_HEADS, eh = (long)E * N_HEADS;
  fill_u32<<<cdiv(nh, 256), 256, 0, stream>>>(mkbuf, 0u, nh);
  fill_u32<<<cdiv(nh, 256), 256, 0, stream>>>((unsigned*)sumbuf, 0u, nh);
  fill_u32<<<cdiv((long)N * HID, 256), 256, 0, stream>>>((unsigned*)aggbuf, 0u, (long)N * HID);

  scores_kernel<<<cdiv(eh, 256), 256, 0, stream>>>(qbuf, kbuf, dst, scbuf, eh);
  segmax_kernel<<<cdiv(eh, 256), 256, 0, stream>>>(scbuf, dst, mkbuf, eh);
  expsum_kernel<<<cdiv(eh, 256), 256, 0, stream>>>(scbuf, dst, mkbuf, escbuf, sumbuf, eh);
  agg_x2h_kernel<<<cdiv((long)E * HID, 256), 256, 0, stream>>>(
      escbuf, sumbuf, vbuf, dst, aggbuf, (long)E * HID);

  nodecat_kernel<<<cdiv((long)N * 256, 256), 256, 0, stream>>>(aggbuf, h, catbuf, (long)N * 256);
  mlp_fused<256, 128><<<ntiles, 128, 0, stream>>>(
      N, 0, catbuf, nullptr, nullptr, nullptr, nullptr, nullptr,
      no_w1t, no_b1, no_g, no_be, no_w2t, no_b2,
      nullptr, /*resid=*/h, newh16, out_h);

  // ============================ h2x layer ============================
  mlp_fused<128, 128><<<ntiles, 128, 0, stream>>>(
      N, 0, newh16, nullptr, nullptr, nullptr, nullptr, nullptr,
      xq_w1t, xq_b1, xq_g, xq_be, xq_w2t, xq_b2,
      nullptr, nullptr, nullptr, qbuf);
  mlp_fused<KVPAD, 128><<<etiles, 128, 0, stream>>>(
      E, 1, nullptr, ea, rfeat16, newh16, src, dst,
      xk_w1t, xk_b1, xk_g, xk_be, xk_w2t, xk_b2,
      nullptr, nullptr, kbuf, nullptr);
  mlp_fused<KVPAD, 16><<<etiles, 128, 0, stream>>>(
      E, 1, nullptr, ea, rfeat16, newh16, src, dst,
      xv_w1t, xv_b1, xv_g, xv_be, xv_w2t, xv_b2,
      ew2, nullptr, vbuf /* v2 [E,16] */, nullptr);

  fill_u32<<<cdiv(nh, 256), 256, 0, stream>>>(mkbuf, 0u, nh);
  fill_u32<<<cdiv(nh, 256), 256, 0, stream>>>((unsigned*)sumbuf, 0u, nh);
  fill_u32<<<cdiv((long)N * 3, 256), 256, 0, stream>>>((unsigned*)delta, 0u, (long)N * 3);

  scores_kernel<<<cdiv(eh, 256), 256, 0, stream>>>(qbuf, kbuf, dst, scbuf, eh);
  segmax_kernel<<<cdiv(eh, 256), 256, 0, stream>>>(scbuf, dst, mkbuf, eh);
  expsum_kernel<<<cdiv(eh, 256), 256, 0, stream>>>(scbuf, dst, mkbuf, escbuf, sumbuf, eh);
  agg_h2x_kernel<<<cdiv(eh, 256), 256, 0, stream>>>(escbuf, sumbuf, vbuf, relx, dst, delta, eh);

  finalx_kernel<<<cdiv((long)N * 3, 256), 256, 0, stream>>>(x, delta, msk, out_x, (long)N * 3);
}
